// RobustGraphAttentionLayer_35991825941096
// MI455X (gfx1250) — matrix operand order repaired and checked by reading, not compiled, not run on hardware
//
#include <hip/hip_runtime.h>

#define NN    8192
#define INF_  512
#define OUTF  64
#define ALPHA 0.2f
#define EPS_  0.01f
#define LOG2E 1.44269504088896f

typedef __attribute__((ext_vector_type(16))) __bf16 v16bf;
typedef __attribute__((ext_vector_type(8)))  float  v8f;

union V16U { uint4 q[2]; v16bf v; };

static __device__ __forceinline__ v16bf ldpair(const unsigned short* p0, const unsigned short* p1) {
  V16U u;
  u.q[0] = *(const uint4*)p0;
  u.q[1] = *(const uint4*)p1;
  return u.v;
}

static __device__ __forceinline__ v16bf zero16() {
  V16U u;
  u.q[0] = uint4{0u,0u,0u,0u};
  u.q[1] = uint4{0u,0u,0u,0u};
  return u.v;
}

static __device__ __forceinline__ v8f bwmma(v16bf a, v16bf b, v8f c) {
  // D = A(16x32 bf16) x B(32x16 bf16) + C(16x16 f32)
  return __builtin_amdgcn_wmma_f32_16x16x32_bf16(false, a, false, b, (short)0, c, false, false);
}

static __device__ __forceinline__ unsigned short f2bf(float f) {
  unsigned u = __builtin_bit_cast(unsigned, f);
  u += 0x7fffu + ((u >> 16) & 1u);          // round-to-nearest-even
  return (unsigned short)(u >> 16);
}

static __device__ __forceinline__ float fast_exp(float x) {
  return __builtin_amdgcn_exp2f(x * LOG2E);
}

// ---------------------------------------------------------------- cast inputs
__global__ void k_cast(const float* __restrict__ x, const float* __restrict__ W,
                       unsigned short* __restrict__ Xb, unsigned short* __restrict__ Wt) {
  int idx = blockIdx.x * 256 + threadIdx.x;
  if (idx < NN * INF_) Xb[idx] = f2bf(x[idx]);
  if (idx < INF_ * OUTF) {                     // W is (512,64); Wt is (64,512)
    int k = idx >> 6, n = idx & 63;
    Wt[n * INF_ + k] = f2bf(W[idx]);
  }
}

// ---------------------------------------------------------------- pack adjacency to bits
__global__ void k_pack(const int* __restrict__ adj, unsigned* __restrict__ bits) {
  size_t w = (size_t)blockIdx.x * 256 + threadIdx.x;       // word id, NN*NN/32 total
  const int* p = adj + w * 32;
  unsigned m = 0;
  #pragma unroll
  for (int k = 0; k < 32; ++k) m |= (p[k] != 0 ? 1u : 0u) << k;
  bits[w] = m;
}

// ---------------------------------------------------------------- h = x @ W  (WMMA bf16)
__global__ __launch_bounds__(256) void k_hgemm(const unsigned short* __restrict__ Xb,
                                               const unsigned short* __restrict__ Wt,
                                               float* __restrict__ hF,
                                               unsigned short* __restrict__ hB,
                                               unsigned short* __restrict__ Vt) {
  int lane = threadIdx.x & 31, wv = threadIdx.x >> 5;
  int h = lane >> 4, c = lane & 15;
  int tile = blockIdx.x * 8 + wv;            // 2048 tiles total
  int mt = tile >> 2, nt = tile & 3;
  v8f acc = {};
  const unsigned short* ap = Xb + (size_t)(mt * 16 + c) * INF_;
  const unsigned short* bp = Wt + (size_t)(nt * 16 + c) * INF_;
  for (int kc = 0; kc < INF_ / 32; ++kc) {
    v16bf A = ldpair(ap + kc * 32 + h * 8, ap + kc * 32 + 16 + h * 8);
    v16bf B = ldpair(bp + kc * 32 + h * 8, bp + kc * 32 + 16 + h * 8);
    acc = bwmma(A, B, acc);
  }
  #pragma unroll
  for (int v = 0; v < 8; ++v) {
    int R = mt * 16 + v + 8 * h, C = nt * 16 + c;
    float val = acc[v];
    hF[(size_t)R * OUTF + C] = val;
    unsigned short bv = f2bf(val);
    hB[(size_t)R * OUTF + C] = bv;
    Vt[(size_t)C * NN + R] = bv;             // transposed copy for B-operand loads
  }
}

// ---------------------------------------------------------------- s1 = h@a1, s2 = h@a2, vnorm = ||h||^2
__global__ void k_sstats(const float* __restrict__ hF, const float* __restrict__ a,
                         float* __restrict__ s1, float* __restrict__ s2, float* __restrict__ vn) {
  int r = blockIdx.x * 256 + threadIdx.x;
  if (r >= NN) return;
  float x1 = 0.f, x2 = 0.f, n2 = 0.f;
  #pragma unroll 8
  for (int c = 0; c < OUTF; ++c) {
    float hv = hF[(size_t)r * OUTF + c];
    x1 += hv * a[c];
    x2 += hv * a[OUTF + c];
    n2 += hv * hv;
  }
  s1[r] = x1; s2[r] = x2; vn[r] = n2;
}

// ---------------------------------------------------------------- per-row softmax stats
__global__ __launch_bounds__(256) void k_rowstats(const unsigned* __restrict__ bits,
                                                  const float* __restrict__ s1,
                                                  const float* __restrict__ s2,
                                                  float* __restrict__ rowmax,
                                                  float* __restrict__ zinv) {
  __shared__ float red[256];
  int r = blockIdx.x, t = threadIdx.x;
  float si = s1[r];
  unsigned word = bits[(size_t)r * (NN / 32) + t];
  float loc[32];
  #pragma unroll
  for (int k = 0; k < 32; ++k) {
    float tv = si + s2[t * 32 + k];
    loc[k] = tv > 0.f ? tv : ALPHA * tv;
  }
  float mx = -3.0e38f;
  #pragma unroll
  for (int k = 0; k < 32; ++k)
    if ((word >> k) & 1u) mx = fmaxf(mx, loc[k]);
  red[t] = mx; __syncthreads();
  for (int s = 128; s > 0; s >>= 1) { if (t < s) red[t] = fmaxf(red[t], red[t + s]); __syncthreads(); }
  float rm = red[0]; __syncthreads();
  float sum = 0.f;
  #pragma unroll
  for (int k = 0; k < 32; ++k)
    if ((word >> k) & 1u) sum += fast_exp(loc[k] - rm);
  red[t] = sum; __syncthreads();
  for (int s = 128; s > 0; s >>= 1) { if (t < s) red[t] += red[t + s]; __syncthreads(); }
  if (t == 0) { rowmax[r] = rm; float Z = red[0]; zinv[r] = Z > 0.f ? 1.f / Z : 0.f; }
}

// ---------------------------------------------------------------- fused robust attention pass
// MODE 0:  M = A @ V           (attention regenerated on the fly)
// MODE 1:  M = (w.*A / rowsum) @ V   with w from distances ||M_i - V_j||
template <int MODE>
__global__ __launch_bounds__(256) void k_mcp(const unsigned short* __restrict__ V,
                                             const unsigned short* __restrict__ Vt,
                                             const unsigned short* __restrict__ Mb,
                                             const float* __restrict__ mnorm,
                                             const float* __restrict__ vnorm,
                                             const float* __restrict__ s1v,
                                             const float* __restrict__ s2v,
                                             const float* __restrict__ rowmax,
                                             const float* __restrict__ zinv,
                                             const unsigned* __restrict__ bits,
                                             float* __restrict__ Mout) {
  __shared__ __align__(16) unsigned short stage[8][512];   // per-wave 16x32 bf16 P tile
  const int lane = threadIdx.x & 31, wv = threadIdx.x >> 5;
  const int h = lane >> 4, c = lane & 15;
  const int rbase = blockIdx.x * 128 + wv * 16;

  float rs1[8], rmx[8], rzi[8], rmn[8];
  #pragma unroll
  for (int v = 0; v < 8; ++v) {
    int r = rbase + v + 8 * h;                 // C-layout row owned by this lane/VGPR
    rs1[v] = s1v[r]; rmx[v] = rowmax[r]; rzi[v] = zinv[r];
    rmn[v] = MODE ? mnorm[r] : 0.f;
  }

  v16bf mA0 = zero16(), mA1 = zero16();        // A-operand: current M rows (16x64, two K-chunks)
  if constexpr (MODE) {
    const unsigned short* mp = Mb + (size_t)(rbase + c) * OUTF;
    mA0 = ldpair(mp + h * 8,      mp + 16 + h * 8);
    mA1 = ldpair(mp + 32 + h * 8, mp + 48 + h * 8);
  }

  v8f num[4] = {};                             // 16x64 f32 accumulator (4 col-tiles)
  float den[8] = {0.f,0.f,0.f,0.f,0.f,0.f,0.f,0.f};

  for (int jb = 0; jb < NN; jb += 32) {
    float S[2][8];
    if constexpr (MODE) {                      // S = M(16x64) @ V_chunk^T(64x32)
      #pragma unroll
      for (int js = 0; js < 2; ++js) {
        const unsigned short* vp = V + (size_t)(jb + js * 16 + c) * OUTF;
        v16bf b0 = ldpair(vp + h * 8,      vp + 16 + h * 8);
        v16bf b1 = ldpair(vp + 32 + h * 8, vp + 48 + h * 8);
        v8f acc = {};
        acc = bwmma(mA0, b0, acc);
        acc = bwmma(mA1, b1, acc);
        #pragma unroll
        for (int v = 0; v < 8; ++v) S[js][v] = acc[v];
      }
    }

    unsigned bw[8];                            // adjacency bits: one 32-wide word per row
    #pragma unroll
    for (int v = 0; v < 8; ++v)
      bw[v] = bits[(size_t)(rbase + v + 8 * h) * (NN / 32) + (jb >> 5)];

    #pragma unroll
    for (int js = 0; js < 2; ++js) {
      int col = jb + js * 16 + c;
      float s2c = s2v[col];
      float vnc = MODE ? vnorm[col] : 0.f;
      #pragma unroll
      for (int v = 0; v < 8; ++v) {
        float t = rs1[v] + s2c;
        float e = t > 0.f ? t : ALPHA * t;
        float Aij = ((bw[v] >> (js * 16 + c)) & 1u) ? fast_exp(e - rmx[v]) * rzi[v] : 0.f;
        float P;
        if constexpr (MODE) {
          float d2   = fmaxf(rmn[v] + vnc - 2.f * S[js][v], 0.f);
          float dist = __builtin_amdgcn_sqrtf(d2);
          float w    = __builtin_amdgcn_rcpf(dist + EPS_) - 0.25f;  // 1/(d+eps) - 1/GAMMA
          w = fmaxf(w, EPS_);
          P = w * Aij;
        } else {
          P = Aij;
        }
        den[v] += P;
        stage[wv][(v + 8 * h) * 32 + js * 16 + c] = f2bf(P);  // row-major 16x32
      }
    }

    asm volatile("s_wait_dscnt 0" ::: "memory");  // P staged in LDS before A-operand reload

    const unsigned short* sp = &stage[wv][c * 32];
    v16bf pA = ldpair(sp + h * 16, sp + h * 16 + 8);

    if (jb + 32 < NN) {
      __builtin_prefetch(V + (size_t)(jb + 32 + c) * OUTF, 0, 1);
      __builtin_prefetch(Vt + (size_t)c * NN + jb + 32, 0, 1);
    }

    #pragma unroll
    for (int nt = 0; nt < 4; ++nt) {           // num += P(16x32) @ V_chunk(32x64)
      const unsigned short* tp = Vt + (size_t)(nt * 16 + c) * NN + jb + h * 16;
      v16bf b = ldpair(tp, tp + 8);
      num[nt] = bwmma(pA, b, num[nt]);
    }
  }

  // row-sum reduction: butterfly within each 16-lane half (wave32)
  #pragma unroll
  for (int v = 0; v < 8; ++v) {
    den[v] += __shfl_xor(den[v], 1, 32);
    den[v] += __shfl_xor(den[v], 2, 32);
    den[v] += __shfl_xor(den[v], 4, 32);
    den[v] += __shfl_xor(den[v], 8, 32);
  }

  #pragma unroll
  for (int nt = 0; nt < 4; ++nt) {
    #pragma unroll
    for (int v = 0; v < 8; ++v) {
      float val = num[nt][v];
      if constexpr (MODE) val = val / fmaxf(den[v], 1e-12f);
      Mout[(size_t)(rbase + v + 8 * h) * OUTF + nt * 16 + c] = val;
    }
  }
}

// ---------------------------------------------------------------- M -> bf16 + row norms
__global__ void k_prep(const float* __restrict__ M, unsigned short* __restrict__ Mb,
                       float* __restrict__ mnorm) {
  int r = blockIdx.x * 256 + threadIdx.x;
  if (r >= NN) return;
  float s = 0.f;
  #pragma unroll 8
  for (int c = 0; c < OUTF; ++c) {
    float v = M[(size_t)r * OUTF + c];
    s += v * v;
    Mb[(size_t)r * OUTF + c] = f2bf(v);
  }
  mnorm[r] = s;
}

// ---------------------------------------------------------------- ELU
__global__ void k_elu(const float* __restrict__ M, float* __restrict__ out) {
  int idx = blockIdx.x * 256 + threadIdx.x;
  if (idx >= NN * OUTF) return;
  float v = M[idx];
  out[idx] = v > 0.f ? v : (fast_exp(v) - 1.f);
}

// ================================================================ host
extern "C" void kernel_launch(void* const* d_in, const int* in_sizes, int n_in,
                              void* d_out, int out_size, void* d_ws, size_t ws_size,
                              hipStream_t stream) {
  (void)in_sizes; (void)n_in; (void)out_size; (void)ws_size;
  const int*   adj = (const int*)d_in[0];
  const float* x   = (const float*)d_in[1];
  const float* W   = (const float*)d_in[2];
  const float* a   = (const float*)d_in[3];
  float*       out = (float*)d_out;

  char* p = (char*)d_ws;
  auto alloc = [&](size_t bytes) -> void* {
    void* r = (void*)p;
    p += (bytes + 255) & ~(size_t)255;
    return r;
  };
  unsigned*       bits = (unsigned*)      alloc((size_t)NN * NN / 8);     // 8 MB bitmask
  unsigned short* Xb   = (unsigned short*)alloc((size_t)NN * INF_ * 2);   // x bf16
  unsigned short* Wt   = (unsigned short*)alloc((size_t)OUTF * INF_ * 2); // W^T bf16
  float*          hF   = (float*)         alloc((size_t)NN * OUTF * 4);   // h f32
  unsigned short* hB   = (unsigned short*)alloc((size_t)NN * OUTF * 2);   // h bf16 (V)
  unsigned short* Vt   = (unsigned short*)alloc((size_t)OUTF * NN * 2);   // h^T bf16
  float*          s1   = (float*)alloc((size_t)NN * 4);
  float*          s2   = (float*)alloc((size_t)NN * 4);
  float*          vn   = (float*)alloc((size_t)NN * 4);
  float*          rmx  = (float*)alloc((size_t)NN * 4);
  float*          zin  = (float*)alloc((size_t)NN * 4);
  float*          mn   = (float*)alloc((size_t)NN * 4);
  float*          M0   = (float*)alloc((size_t)NN * OUTF * 4);
  float*          M1   = (float*)alloc((size_t)NN * OUTF * 4);
  unsigned short* Mb   = (unsigned short*)alloc((size_t)NN * OUTF * 2);

  k_cast<<<(NN * INF_ + 255) / 256, 256, 0, stream>>>(x, W, Xb, Wt);
  k_pack<<<(NN * NN / 32) / 256, 256, 0, stream>>>(adj, bits);
  k_hgemm<<<(NN / 16) * (OUTF / 16) / 8, 256, 0, stream>>>(Xb, Wt, hF, hB, Vt);
  k_sstats<<<NN / 256, 256, 0, stream>>>(hF, a, s1, s2, vn);
  k_rowstats<<<NN, 256, 0, stream>>>(bits, s1, s2, rmx, zin);

  // M0 = A @ V   (attention regenerated on the fly)
  k_mcp<0><<<NN / 128, 256, 0, stream>>>(hB, Vt, Mb, mn, vn, s1, s2, rmx, zin, bits, M0);

  float* cur = M0;
  float* nxt = M1;
  for (int it = 0; it < 3; ++it) {
    k_prep<<<NN / 256, 256, 0, stream>>>(cur, Mb, mn);
    k_mcp<1><<<NN / 128, 256, 0, stream>>>(hB, Vt, Mb, mn, vn, s1, s2, rmx, zin, bits, nxt);
    float* t = cur; cur = nxt; nxt = t;
  }

  k_elu<<<(NN * OUTF + 255) / 256, 256, 0, stream>>>(cur, out);
}
